// QuantLinearLUT_1039382086454
// MI455X (gfx1250) — matrix operand order repaired and checked by reading, not compiled, not run on hardware
//
#include <hip/hip_runtime.h>
#include <hip/hip_bf16.h>

// ---------------------------------------------------------------------------
// QuantLinearLUT (SqueezeLLM-style) for MI455X / gfx1250
//   y[8192,4096] = x[8192,4096] @ W^T + bias
//   W[4096,4096] = LUT[o][nibble(qweight)] + CSR outliers
//
// Phase 1: dequant W -> bf16 workspace (32 MB, L2-resident)
// Phase 2: bf16 WMMA GEMM (v_wmma_f32_16x16x32_bf16), f32 accumulate.
//          B (weight) tiles stream via GLOBAL_LOAD_ASYNC_TO_LDS_B128
//          (ASYNCcnt + s_wait_asynccnt), A tiles register-staged with
//          on-the-fly f32->bf16 conversion. Double-buffered LDS.
// ---------------------------------------------------------------------------

typedef __bf16 bf16_t;
typedef __attribute__((ext_vector_type(8)))  __bf16 v8bf;
typedef __attribute__((ext_vector_type(16))) __bf16 v16bf;
typedef __attribute__((ext_vector_type(8)))  float  v8f;

#define IN_F   4096
#define OUT_F  4096
#define TM     128
#define TN     128
#define TK     32
#define LDT    48          // LDS row pitch in halves (96B: multiple of 32B)
#define NTHREADS 256

static __device__ __forceinline__ v16bf concat8(v8bf lo, v8bf hi) {
    return __builtin_shufflevector(lo, hi, 0,1,2,3,4,5,6,7,8,9,10,11,12,13,14,15);
}

// Truncate a flat pointer to the 32-bit LDS byte address (LDS aperture lives
// entirely in the upper 32 bits; ISA 10.2 aperture mapping).
static __device__ __forceinline__ unsigned lds_addr32(const void* p) {
    return (unsigned)(size_t)p;
}

// ---------------------------------------------------------------------------
// Phase 1: one block per output row o.
// ---------------------------------------------------------------------------
__global__ __launch_bounds__(NTHREADS)
void dequant_rows(const int* __restrict__ qweight,      // [IN/8, OUT]
                  const float* __restrict__ lut,        // [OUT, 16]
                  const float* __restrict__ ovals,      // [NNZ]
                  const int* __restrict__ orows,        // [OUT+1]
                  const int* __restrict__ ocols,        // [NNZ]
                  bf16_t* __restrict__ W16)             // [OUT, IN]
{
    __shared__ float row[IN_F];
    const int o = blockIdx.x;
    const int t = threadIdx.x;

    // per-output-channel LUT (uniform across block -> scalar loads)
    float l[16];
#pragma unroll
    for (int j = 0; j < 16; ++j) l[j] = lut[o * 16 + j];

    // unpack packed nibbles: qweight[w][o], 8 indices per word
    for (int w = t; w < IN_F / 8; w += NTHREADS) {
        unsigned q = (unsigned)qweight[(size_t)w * OUT_F + o];
#pragma unroll
        for (int j = 0; j < 8; ++j)
            row[w * 8 + j] = l[(q >> (4 * j)) & 0xF];
    }
    __syncthreads();

    // CSR outliers for row o (ds_add_f32 atomics; columns may repeat)
    const int s = orows[o], e = orows[o + 1];
    for (int j = s + t; j < e; j += NTHREADS)
        atomicAdd(&row[ocols[j] & (IN_F - 1)], ovals[j]);
    __syncthreads();

    // convert + store: each thread writes 16 contiguous bf16 (2 x 16B)
    const int i = t * 16;
    v8bf p0, p1;
#pragma unroll
    for (int j = 0; j < 8; ++j) {
        p0[j] = (bf16_t)row[i + j];
        p1[j] = (bf16_t)row[i + 8 + j];
    }
    bf16_t* dst = W16 + (size_t)o * IN_F + i;
    *(v8bf*)(dst)     = p0;
    *(v8bf*)(dst + 8) = p1;
}

// ---------------------------------------------------------------------------
// Phase 2: tiled WMMA GEMM, double-buffered LDS, async B staging.
//   grid = (OUT/TN, B/TM), 256 threads = 8 waves (wave32).
//   Wave (wm in 0..3, wn in 0..1) owns a 32x64 sub-tile:
//     2 (M) x 4 (N) accumulators of 16x16 f32.
// ---------------------------------------------------------------------------
__global__ __launch_bounds__(NTHREADS)
void gemm_wmma_bf16(const float* __restrict__ x,        // [B, IN] f32
                    const bf16_t* __restrict__ W16,     // [OUT, IN] bf16
                    const float* __restrict__ bias,     // [OUT]
                    float* __restrict__ y)              // [B, OUT]
{
    __shared__ bf16_t sA[2][TM * LDT];   // x tile,  [m][k], bf16
    __shared__ bf16_t sB[2][TN * LDT];   // W tile,  [n][k], bf16

    const int tid  = threadIdx.x;
    const int lane = tid & 31;
    const int wave = tid >> 5;
    const int rl   = lane & 15;      // row-in-subtile
    const int rs   = lane >> 4;      // K-half selector (per ISA A/B lane layout)

    const int bm = blockIdx.y * TM;
    const int bn = blockIdx.x * TN;
    const int wm = (wave & 3) * 32;  // wave M offset in block tile
    const int wn = (wave >> 2) * 64; // wave N offset in block tile

    // Staging map: tile = 128 rows x 32 elems = 512 chunks of 8 elems.
    // Thread t handles chunks t and t+256.
    int rA[2], cA[2];
#pragma unroll
    for (int u = 0; u < 2; ++u) {
        const int c = tid + u * NTHREADS;
        rA[u] = c >> 2;
        cA[u] = (c & 3) * 8;
    }

    // --- helpers -----------------------------------------------------------
    v8f ra[2];                               // A chunk (8 f32) in regs
    auto load_a_regs = [&](int kt) {
        const int k0 = kt * TK;
#pragma unroll
        for (int u = 0; u < 2; ++u)
            ra[u] = *(const v8f*)(x + (size_t)(bm + rA[u]) * IN_F + k0 + cA[u]);
    };
    auto commit_a = [&](int buf) {
#pragma unroll
        for (int u = 0; u < 2; ++u) {
            v8bf ha;
#pragma unroll
            for (int j = 0; j < 8; ++j) ha[j] = (bf16_t)ra[u][j];
            *(v8bf*)(&sA[buf][rA[u] * LDT + cA[u]]) = ha;
        }
    };
    // B tile: global bf16 -> LDS directly, no VGPR staging (ASYNCcnt).
    auto async_b = [&](int kt, int buf) {
        const int k0 = kt * TK;
#pragma unroll
        for (int u = 0; u < 2; ++u) {
            unsigned long long g =
                (unsigned long long)(size_t)(W16 + (size_t)(bn + rA[u]) * IN_F + k0 + cA[u]);
            unsigned l = lds_addr32(&sB[buf][rA[u] * LDT + cA[u]]);
            asm volatile("global_load_async_to_lds_b128 %0, %1, off"
                         :: "v"(l), "v"(g) : "memory");
        }
    };
    // -----------------------------------------------------------------------

    // prologue: stage K-slab 0
    load_a_regs(0);
    async_b(0, 0);

    v8f acc[8];
#pragma unroll
    for (int i = 0; i < 8; ++i) acc[i] = (v8f)0.0f;

    const int KT = IN_F / TK;        // 128 K-steps
    for (int kt = 0; kt < KT; ++kt) {
        const int cur = kt & 1;

        commit_a(cur);                                   // ds_store A regs
        asm volatile("s_wait_asynccnt 0x0" ::: "memory"); // our B async landed
        __syncthreads();                                  // tile published

        // stage next K-slab while computing this one
        if (kt + 1 < KT) {
            async_b(kt + 1, cur ^ 1);
            load_a_regs(kt + 1);
            if (kt + 2 < KT)  // warm near cache two slabs ahead
                __builtin_prefetch(x + (size_t)(bm + rA[0]) * IN_F + (kt + 2) * TK, 0, 3);
        }

        // ---- fragments per ISA 16-bit layouts ----
        // A 16x32: lane rl holds row M=rl; lanes 0-15 carry K {0-7,16-23},
        //          lanes 16-31 carry K {8-15,24-31}.
        v16bf afrag[2];
#pragma unroll
        for (int i = 0; i < 2; ++i) {
            const bf16_t* base = &sA[cur][(wm + i * 16 + rl) * LDT];
            v8bf lo = *(const v8bf*)(base + rs * 8);
            v8bf hi = *(const v8bf*)(base + 16 + rs * 8);
            afrag[i] = concat8(lo, hi);
        }
        // B 32x16 (K x N), B[k][n] = W[bn+n][k]: lane rl holds column N=rl;
        // lanes 0-15 carry K {0-7,16-23}, lanes 16-31 carry K {8-15,24-31}.
        v16bf bfrag[4];
#pragma unroll
        for (int j = 0; j < 4; ++j) {
            const bf16_t* base = &sB[cur][(wn + j * 16 + rl) * LDT + rs * 8];
            v8bf lo = *(const v8bf*)(base);
            v8bf hi = *(const v8bf*)(base + 16);
            bfrag[j] = concat8(lo, hi);
        }

        // 8 WMMAs: D = A x B + C  (f32 accumulate)
#pragma unroll
        for (int i = 0; i < 2; ++i)
#pragma unroll
            for (int j = 0; j < 4; ++j)
                acc[i * 4 + j] = __builtin_amdgcn_wmma_f32_16x16x32_bf16(
                    /*neg_a=*/false, afrag[i],
                    /*neg_b=*/false, bfrag[j],
                    /*c_mod=*/(short)0, acc[i * 4 + j],
                    /*reuse_a=*/false, /*reuse_b=*/false);
    }

    // ---- epilogue: C layout = VGPR r, lanes 0-15 -> M=r/N=lane,
    //                lanes 16-31 -> M=r+8/N=lane-16 ----
#pragma unroll
    for (int j = 0; j < 4; ++j) {
        const int n  = bn + wn + j * 16 + rl;
        const float bv = bias[n];
#pragma unroll
        for (int i = 0; i < 2; ++i) {
            const int mbase = bm + wm + i * 16 + rs * 8;
#pragma unroll
            for (int r = 0; r < 8; ++r)
                y[(size_t)(mbase + r) * OUT_F + n] = acc[i * 4 + j][r] + bv;
        }
    }
}

// ---------------------------------------------------------------------------
extern "C" void kernel_launch(void* const* d_in, const int* in_sizes, int n_in,
                              void* d_out, int out_size, void* d_ws, size_t ws_size,
                              hipStream_t stream)
{
    const float* x     = (const float*)d_in[0];   // [4,2048,4096]
    const float* lut   = (const float*)d_in[1];   // [4096,16]
    const float* bias  = (const float*)d_in[2];   // [4096]
    const float* ovals = (const float*)d_in[3];   // [NNZ]
    const int*   qw    = (const int*)  d_in[4];   // [512,4096]
    const int*   orows = (const int*)  d_in[5];   // [4097]
    const int*   ocols = (const int*)  d_in[6];   // [NNZ]
    float*       y     = (float*)d_out;

    bf16_t* W16 = (bf16_t*)d_ws;                  // 4096*4096*2 = 32 MB scratch
    const int Brows = in_sizes[0] / IN_F;         // 8192

    // Phase 1: dequant + outlier correction -> bf16 W
    dequant_rows<<<OUT_F, NTHREADS, 0, stream>>>(qw, lut, ovals, orows, ocols, W16);

    // Phase 2: WMMA GEMM
    dim3 grid(OUT_F / TN, Brows / TM);            // (32, 64)
    gemm_wmma_bf16<<<grid, NTHREADS, 0, stream>>>(x, W16, bias, y);
}
